// ChannelMixingConv1D_36232344109481
// MI455X (gfx1250) — matrix operand order repaired and checked by reading, not compiled, not run on hardware
//
#include <hip/hip_runtime.h>
#include <hip/hip_bf16.h>

typedef __bf16 v16bf __attribute__((ext_vector_type(16)));
typedef __bf16 v8bf  __attribute__((ext_vector_type(8)));
typedef float  v8f   __attribute__((ext_vector_type(8)));

#define L_LEN 2048
#define CIN   128
#define COUT  256
#define BATCH 64
#define EPS   1e-5f

__device__ __forceinline__ unsigned short f2bf(float f) {
    unsigned int u = __float_as_uint(f);
    unsigned int r = u + 0x7FFFu + ((u >> 16) & 1u);
    return (unsigned short)(r >> 16);
}

// ---------------------------------------------------------------------------
// Kernel 0: convert mix_w [256,128] fp32 -> bf16 into ws (L2-resident operand)
// ---------------------------------------------------------------------------
__global__ __launch_bounds__(256) void k_prep(const float* __restrict__ mix_w,
                                              unsigned short* __restrict__ mwb) {
    int i = blockIdx.x * 256 + threadIdx.x;
    if (i < COUT * CIN) mwb[i] = f2bf(mix_w[i]);
}

// ---------------------------------------------------------------------------
// Kernel 1: depthwise conv (k=3,pad=1) -> bf16 y tile in LDS, then WMMA GEMM.
// Block = 256 threads (8 waves), owns 128 contiguous l-columns of one batch.
// Wave w owns M rows [32w, 32w+32) (2 M-tiles); its 8 A-fragments are loaded
// once from L2-resident bf16 mix_w and held in VGPRs; the n-loop streams
// B-fragments from LDS and issues 8 WMMAs per iteration.
// ---------------------------------------------------------------------------
__global__ __launch_bounds__(256) void k_dwconv_gemm(
        const float* __restrict__ x,
        const float* __restrict__ dw_w,
        const float* __restrict__ dw_b,
        const float* __restrict__ mix_b,
        const unsigned short* __restrict__ mwb,   // bf16 mix_w [256][128]
        float* __restrict__ z) {                  // d_out used as z scratch
    __shared__ alignas(16) unsigned short y_lds[128 * 136];  // [col j][chan c]

    const int tid  = threadIdx.x;
    const int wave = tid >> 5;
    const int lane = tid & 31;
    const int n0   = blockIdx.x << 7;        // 128 columns per block
    const int b    = n0 >> 11;               // n0 / 2048
    const int l0   = n0 & (L_LEN - 1);

    // ---- Phase 1: depthwise conv into LDS as bf16 ----
    for (int i = tid; i < 128 * CIN; i += 256) {
        int c = i >> 7;
        int j = i & 127;
        int l = l0 + j;
        const float* xr = x + (((size_t)b * CIN + c) << 11);
        float w0 = dw_w[3 * c], w1 = dw_w[3 * c + 1], w2 = dw_w[3 * c + 2];
        float xm = (l > 0)         ? xr[l - 1] : 0.0f;
        float xc = xr[l];
        float xp = (l < L_LEN - 1) ? xr[l + 1] : 0.0f;
        float yv = fmaf(w0, xm, fmaf(w1, xc, fmaf(w2, xp, dw_b[c])));
        y_lds[j * 136 + c] = f2bf(yv);
    }

    // ---- A-fragments: 2 M-tiles x 4 K-chunks, loaded once (overlaps Phase 1) ----
    // A layout: lane L -> row m = L&15 ; elems 0..7 -> K=kb8+e, 8..15 -> K=kb8+16+e
    const int ma  = lane & 15;
    const int hi  = lane >> 4;               // 0 or 1
    const int kb8 = hi << 3;
    v16bf afrag[2][4];
#pragma unroll
    for (int t = 0; t < 2; ++t) {
        const int m0 = ((wave << 1) + t) << 4;
#pragma unroll
        for (int kc = 0; kc < 4; ++kc) {
            const unsigned short* ap = mwb + (m0 + ma) * CIN + (kc << 5) + kb8;
            ((v8bf*)&afrag[t][kc])[0] = *(const v8bf*)ap;         // K = kb8..kb8+7
            ((v8bf*)&afrag[t][kc])[1] = *(const v8bf*)(ap + 16);  // K = kb8+16..
        }
    }

    // Bias per C-row: D layout lane L, vgpr r -> m = m0 + r + 8*hi
    const int m_base = (wave << 5) + (hi << 3);   // wave*32 + 8*hi
    float bias0[8], bias1[8];
#pragma unroll
    for (int r = 0; r < 8; ++r) {
        bias0[r] = mix_b[m_base + r];
        bias1[r] = mix_b[m_base + 16 + r];
    }
    __syncthreads();

    // ---- Phase 2: loop the 8 N-tiles ----
    const int kb16 = hi << 3;                // B layout: lane L -> n=L&15, K=(L>>4)*8+e, +16 for e>=8
    const int ncol = lane & 15;
    for (int nt = 0; nt < 8; ++nt) {
        const int jn = (nt << 4) + ncol;
        v16bf bfr[4];
#pragma unroll
        for (int kc = 0; kc < 4; ++kc) {
            const v8bf* p = (const v8bf*)&y_lds[jn * 136 + (kc << 5) + kb16];
            ((v8bf*)&bfr[kc])[0] = p[0];
            ((v8bf*)&bfr[kc])[1] = p[2];
        }
        v8f acc0 = {0.f, 0.f, 0.f, 0.f, 0.f, 0.f, 0.f, 0.f};
        v8f acc1 = {0.f, 0.f, 0.f, 0.f, 0.f, 0.f, 0.f, 0.f};
#pragma unroll
        for (int kc = 0; kc < 4; ++kc) {
            acc0 = __builtin_amdgcn_wmma_f32_16x16x32_bf16(
                false, afrag[0][kc], false, bfr[kc], (short)0, acc0, false, false);
            acc1 = __builtin_amdgcn_wmma_f32_16x16x32_bf16(
                false, afrag[1][kc], false, bfr[kc], (short)0, acc1, false, false);
        }
        // Epilogue: bias + store (lanes 0-15 write 64B-coalesced segments)
        const int lg = l0 + jn;
        float* zp = z + (((size_t)b * COUT + m_base) << 11) + lg;
#pragma unroll
        for (int r = 0; r < 8; ++r) {
            zp[(size_t)r << 11]        = acc0[r] + bias0[r];
            zp[(size_t)(r + 16) << 11] = acc1[r] + bias1[r];
        }
    }
}

// ---------------------------------------------------------------------------
// Kernel 2: streaming per-channel partial sum / sumsq over z.
// Grid = 256 channels x 4 splits; each block reduces 16 batches of its channel.
// ---------------------------------------------------------------------------
__global__ __launch_bounds__(256) void k_reduce(const float* __restrict__ z,
                                                float* __restrict__ psum,
                                                float* __restrict__ psq) {
    const int m    = blockIdx.x >> 2;
    const int part = blockIdx.x & 3;
    const int tid  = threadIdx.x;
    const float4* zp = (const float4*)z;

    float s = 0.0f, s2 = 0.0f;
    // 16 batches x 512 float4 per (batch, channel) row
    for (int i = tid; i < 16 * 512; i += 256) {
        int bb  = (part << 4) + (i >> 9);
        int off = i & 511;
        const float4* p = &zp[((size_t)(bb * COUT + m) << 9) + off];
        __builtin_prefetch(p + 256, 0, 1);     // global_prefetch_b8 a stride ahead
        float4 v = *p;
        s  += v.x + v.y + v.z + v.w;
        s2 += v.x * v.x + v.y * v.y + v.z * v.z + v.w * v.w;
    }
#pragma unroll
    for (int o = 16; o; o >>= 1) {
        s  += __shfl_xor(s, o, 32);
        s2 += __shfl_xor(s2, o, 32);
    }
    __shared__ float rs[8], rq[8];
    if ((tid & 31) == 0) { rs[tid >> 5] = s; rq[tid >> 5] = s2; }
    __syncthreads();
    if (tid == 0) {
        float S = 0.0f, Q = 0.0f;
#pragma unroll
        for (int i = 0; i < 8; ++i) { S += rs[i]; Q += rq[i]; }
        psum[(m << 2) + part] = S;
        psq [(m << 2) + part] = Q;
    }
}

// ---------------------------------------------------------------------------
// Kernel 3: fold partials -> per-channel scale/shift
// ---------------------------------------------------------------------------
__global__ __launch_bounds__(256) void k_stats(const float* __restrict__ psum,
                                               const float* __restrict__ psq,
                                               const float* __restrict__ gamma,
                                               const float* __restrict__ beta,
                                               float* __restrict__ scale,
                                               float* __restrict__ shift) {
    int c = threadIdx.x;
    float S = psum[4 * c] + psum[4 * c + 1] + psum[4 * c + 2] + psum[4 * c + 3];
    float Q = psq [4 * c] + psq [4 * c + 1] + psq [4 * c + 2] + psq [4 * c + 3];
    const float inv_n = 1.0f / ((float)BATCH * (float)L_LEN);
    float mean = S * inv_n;
    float var  = Q * inv_n - mean * mean;
    float rstd = rsqrtf(var + EPS);
    float sc   = gamma[c] * rstd;
    scale[c] = sc;
    shift[c] = fmaf(-mean, sc, beta[c]);
}

// ---------------------------------------------------------------------------
// Kernel 4: in-place normalize + ReLU over d_out, float4 vectorized
// ---------------------------------------------------------------------------
__global__ __launch_bounds__(256) void k_norm_relu(float* __restrict__ z,
                                                   const float* __restrict__ scale,
                                                   const float* __restrict__ shift,
                                                   int n4) {
    int i4 = blockIdx.x * 256 + threadIdx.x;
    if (i4 >= n4) return;
    int c = (i4 >> 9) & (COUT - 1);       // 512 float4 per (b, channel) row
    float sc = scale[c], sh = shift[c];
    float4 v = ((const float4*)z)[i4];
    v.x = fmaxf(fmaf(v.x, sc, sh), 0.0f);
    v.y = fmaxf(fmaf(v.y, sc, sh), 0.0f);
    v.z = fmaxf(fmaf(v.z, sc, sh), 0.0f);
    v.w = fmaxf(fmaf(v.w, sc, sh), 0.0f);
    ((float4*)z)[i4] = v;
}

// ---------------------------------------------------------------------------
extern "C" void kernel_launch(void* const* d_in, const int* in_sizes, int n_in,
                              void* d_out, int out_size, void* d_ws, size_t ws_size,
                              hipStream_t stream) {
    const float* x     = (const float*)d_in[0];
    const float* dw_w  = (const float*)d_in[1];
    const float* dw_b  = (const float*)d_in[2];
    const float* mix_w = (const float*)d_in[3];
    const float* mix_b = (const float*)d_in[4];
    const float* gamma = (const float*)d_in[5];
    const float* beta  = (const float*)d_in[6];

    float* z   = (float*)d_out;            // z lives in d_out, normalized in place
    float* wsf = (float*)d_ws;
    float* psum  = wsf;                    // [1024]
    float* psq   = wsf + 1024;             // [1024]
    float* scale = wsf + 2048;             // [256]
    float* shift = wsf + 2304;             // [256]
    unsigned short* mwb = (unsigned short*)(wsf + 2560);  // bf16 mix_w, 64 KB

    k_prep<<<128, 256, 0, stream>>>(mix_w, mwb);
    k_dwconv_gemm<<<(BATCH * L_LEN) / 128, 256, 0, stream>>>(
        x, dw_w, dw_b, mix_b, mwb, z);
    k_reduce<<<COUT * 4, 256, 0, stream>>>(z, psum, psq);
    k_stats<<<1, 256, 0, stream>>>(psum, psq, gamma, beta, scale, shift);
    int n4 = (BATCH * COUT * L_LEN) / 4;
    k_norm_relu<<<(n4 + 255) / 256, 256, 0, stream>>>(z, scale, shift, n4);
}
